// AutoBANTModel1d_29231547416817
// MI455X (gfx1250) — compile-verified, matrix operand order checked by reading, not run on hardware
//
#include <hip/hip_runtime.h>

// ---------------------------------------------------------------------------
// AutoBANTModel1d forward on gfx1250 (MI455X).
// bf16 activations/weights, f32 accumulation via v_wmma_f32_16x16x32_bf16.
// BN folded into conv weights/bias; ReLU / *S / residual fused in epilogue.
// The 16 x Kpad weight panel of each output tile is DMA'd once into LDS by
// the Tensor Data Mover (tensor_load_to_lds + s_wait_tensorcnt); the K-loop
// is then barrier-free: 16 immediate-offset B gathers + 2 ds_load_b128 + WMMA.
// ---------------------------------------------------------------------------

typedef __attribute__((ext_vector_type(16))) __bf16 v16bf;
typedef __attribute__((ext_vector_type(8)))  __bf16 v8bf;
typedef __attribute__((ext_vector_type(8)))  float  v8f;
typedef __attribute__((ext_vector_type(4)))  unsigned int v4u;
typedef __attribute__((ext_vector_type(8)))  int v8i;
typedef __attribute__((ext_vector_type(4)))  int v4i;

#define BF16(x) ((__bf16)(x))
#define HALO 8

// ---- S = sum(trust_scores) on device (graph-capture safe) -----------------
__global__ void sum_ts_kernel(const float* __restrict__ ts, float* __restrict__ Sp) {
  if (threadIdx.x == 0 && blockIdx.x == 0) {
    float s = 0.f;
    for (int n = 0; n < 8; ++n) s += ts[n];
    *Sp = s;
  }
}

// ---- f32 -> bf16 conversion into halo-padded rows -------------------------
__global__ void cvt_pad_kernel(const float* __restrict__ in, __bf16* __restrict__ out,
                               long rows, int L, int Lp) {
  long i = (long)blockIdx.x * blockDim.x + threadIdx.x;
  if (i >= rows * (long)L) return;
  long r = i / L;
  int  c = (int)(i % L);
  out[r * (size_t)Lp + HALO + c] = BF16(in[i]);
}

// ---- zero the 2x8-element halos of each padded row ------------------------
__global__ void zero_halo_kernel(__bf16* __restrict__ buf, long rows, int Lp) {
  long i = (long)blockIdx.x * blockDim.x + threadIdx.x;
  if (i >= rows * 16) return;
  long r = i >> 4;
  int  j = (int)(i & 15);
  int  pos = (j < 8) ? j : (Lp - 16 + j);
  buf[r * (size_t)Lp + pos] = BF16(0.f);
}

// ---- trust-weighted conv fusion + BN folding (kk-major K packing) ---------
// wout[o, ik] for ik = kk*cin + ci  <-  bnscale[o] * sum_n ts_n * W[n,o,ci,kk]
// (zero for ik >= cin*kw);  b_eff[o] = bnscale[o]*sum_n ts_n*b[n,o] + bnshift[o]
__global__ void fuse_conv_kernel(const float* __restrict__ w, const float* __restrict__ b,
                                 const float* __restrict__ g, const float* __restrict__ be,
                                 const float* __restrict__ mu, const float* __restrict__ va,
                                 const float* __restrict__ ts,
                                 __bf16* __restrict__ wout, float* __restrict__ bout,
                                 int cout, int cin, int kw, int kpad) {
  int i = blockIdx.x * blockDim.x + threadIdx.x;
  int cink = cin * kw;
  int totp = cout * kpad;
  int tot  = cout * cink;
  if (i < totp) {
    int o  = i / kpad;
    int ik = i - o * kpad;
    float v = 0.f;
    if (ik < cink) {
      int ci = ik % cin;            // kk-major K ordering
      int kk = ik / cin;
      float scale = 0.f, acc = 0.f;
      for (int n = 0; n < 8; ++n) {
        float inv = g[n * cout + o] * rsqrtf(va[n * cout + o] + 1e-5f);
        scale += ts[n] * inv;
        acc   += ts[n] * w[(size_t)n * tot + o * cink + ci * kw + kk];
      }
      v = scale * acc;
    }
    wout[i] = BF16(v);
  }
  if (i < cout) {
    float scale = 0.f, shift = 0.f, bb = 0.f;
    for (int n = 0; n < 8; ++n) {
      float inv = g[n * cout + i] * rsqrtf(va[n * cout + i] + 1e-5f);
      scale += ts[n] * inv;
      shift += ts[n] * (be[n * cout + i] - mu[n * cout + i] * inv);
      bb    += ts[n] * b[n * cout + i];
    }
    bout[i] = scale * bb + shift;
  }
}

// ---- head fusion: linear weights + logits BN affine -----------------------
__global__ void fuse_head_kernel(const float* __restrict__ lw, const float* __restrict__ lb,
                                 const float* __restrict__ g, const float* __restrict__ be,
                                 const float* __restrict__ mu, const float* __restrict__ va,
                                 const float* __restrict__ ts,
                                 float* __restrict__ lwout, float* __restrict__ lbout,
                                 float* __restrict__ hs, float* __restrict__ hh,
                                 int O, int D) {
  int i = blockIdx.x * blockDim.x + threadIdx.x;
  int tot = O * D;
  if (i < tot) {
    float a = 0.f;
    for (int n = 0; n < 8; ++n) a += ts[n] * lw[(size_t)n * tot + i];
    lwout[i] = a;
  }
  if (i < O) {
    float sc = 0.f, sh = 0.f, sb = 0.f;
    for (int n = 0; n < 8; ++n) {
      float inv = g[n * O + i] * rsqrtf(va[n * O + i] + 1e-5f);
      sc += ts[n] * inv;
      sh += ts[n] * (be[n * O + i] - mu[n * O + i] * inv);
      sb += ts[n] * lb[n * O + i];
    }
    lbout[i] = sb; hs[i] = sc; hh[i] = sh;
  }
}

// ---------------------------------------------------------------------------
// Implicit-GEMM conv1d via WMMA bf16.  M = Cout tile 16, N = 16 output
// positions per wave, K = Cin*KW (kk-major) swept in 32-chunks.  Block =
// 4 waves sharing one TDM-staged LDS weight panel over 64 output positions.
// ---------------------------------------------------------------------------
template <int CIN, int KW, int LINP>
__global__ __launch_bounds__(128)
void conv1d_wmma_kernel(const __bf16* __restrict__ in, __bf16* __restrict__ out,
                        const __bf16* __restrict__ wt, const float* __restrict__ bias,
                        const __bf16* __restrict__ resid,
                        int Cout, int LoutP, int stride, int pad,
                        int doRelu, int doS, const float* __restrict__ Sp) {
  constexpr int Ktot = CIN * KW;
  constexpr int Kpad = (Ktot + 31) & ~31;
  constexpr bool POW2 = (CIN >= 32) && ((CIN & (CIN - 1)) == 0);

  __shared__ __attribute__((aligned(16))) __bf16 ldsW[16 * Kpad];

  const int lane  = threadIdx.x & 31;
  const int wv    = threadIdx.x >> 5;
  const int bIdx  = blockIdx.z;
  const int mBase = blockIdx.y * 16;
  const int n     = (blockIdx.x * 4 + wv) * 16 + (lane & 15);
  const int kgrp  = lane >> 4;   // 0: K {0..7,16..23}   1: K {8..15,24..31}

  // ---- stage the whole 16 x Kpad weight panel (contiguous in wq) to LDS ----
#if __has_builtin(__builtin_amdgcn_tensor_load_to_lds)
  if (threadIdx.x < 32) {  // wave 0 issues the TDM DMA (EXEC-independent)
    const unsigned lds_off = (unsigned)(unsigned long long)&ldsW[0];  // LDS byte addr
    const unsigned long long ga =
        (unsigned long long)(wt + (size_t)mBase * Kpad);              // tile base
    // D# group 0: count=1 | lds_addr | global_addr(57b) | type=2
    v4u g0 = { 1u,
               lds_off,
               (unsigned)(ga & 0xFFFFFFFFu),
               (unsigned)((ga >> 32) & 0x01FFFFFFu) | (2u << 30) };
    // D# group 1: data_size=2B; tensor 16 x Kpad; tile 16 x Kpad; stride Kpad
    v8i g1 = { (int)0x00010000,                 // data_size=1 (2 bytes)
               (int)((unsigned)Kpad << 16),     // tensor_dim0[15:0] @ [63:48]
               (int)(16u << 16),                // tensor_dim0 hi=0 | tensor_dim1=16
               (int)((unsigned)Kpad << 16),     // tensor_dim1 hi=0 | tile_dim0=Kpad
               16,                              // tile_dim1=16, tile_dim2=0
               Kpad,                            // tensor_dim0_stride[31:0]
               0, 0 };
    v4i g2 = { 0, 0, 0, 0 };                    // 2D: tile_dim3=0 (unused)
    v4i g3 = { 0, 0, 0, 0 };
    v8i g4 = { 0, 0, 0, 0, 0, 0, 0, 0 };        // extra group (this toolchain's 6-arg form)
    __builtin_amdgcn_tensor_load_to_lds(g0, g1, g2, g3, g4, 0);
    __builtin_amdgcn_s_wait_tensorcnt(0);
  }
#else
  // fallback: cooperative linear copy (panel is contiguous)
  for (int t = threadIdx.x * 4; t < 16 * Kpad; t += 128 * 4)
    *(uint64_t*)&ldsW[t] = *(const uint64_t*)&wt[(size_t)mBase * Kpad + t];
#endif
  __syncthreads();

  // per-lane gather base (halo makes every access in-bounds);
  // fold the kgrp-dependent +8 channel rows in once.
  const __bf16* inp0 = in + (size_t)bIdx * CIN * LINP + n * stride - pad + HALO;
  const __bf16* inpG = inp0 + (kgrp ? 8 * LINP : 0);
  if (resid) __builtin_prefetch(resid + ((size_t)bIdx * Cout + mBase) * LoutP, 0, 1);

  const __bf16* lrow = &ldsW[(lane & 15) * Kpad + (kgrp << 3)];

  v8f acc = {};

#pragma unroll 2
  for (int kb = 0; kb < Kpad; kb += 32) {
    // A fragment: two aligned 16B LDS reads, concat into v16bf
    v8bf a0 = *(const v8bf*)(lrow + kb);
    v8bf a1 = *(const v8bf*)(lrow + kb + 16);
    v16bf av = __builtin_shufflevector(a0, a1, 0, 1, 2, 3, 4, 5, 6, 7,
                                       8, 9, 10, 11, 12, 13, 14, 15);

    // B fragment
    v16bf bv;
    if constexpr (POW2) {
      // chunk never crosses a CIN boundary: kk and ciBase uniform scalars,
      // per-element offset klb*LINP is a compile-time immediate.
      const int ciBase = kb & (CIN - 1);
      const int kk     = kb / CIN;
      const __bf16* pc = inpG + (ciBase * LINP + kk);   // one bump per chunk
#pragma unroll
      for (int e = 0; e < 16; ++e) {
        const int klb = (e & 7) + ((e >> 3) << 4);      // constant per e
        bv[e] = pc[klb * LINP];
      }
    } else {
      // stem only (CIN=12): per-element div/mod by constant, 3 chunks total
#pragma unroll
      for (int e = 0; e < 16; ++e) {
        int kl = (e & 7) + ((e >> 3) << 4) + (kgrp << 3);
        int kg = kb + kl;
        if (kg > Ktot - 1) kg = Ktot - 1;               // pad region: W==0
        int ci = kg % CIN;
        int kk = kg / CIN;
        bv[e] = inp0[ci * LINP + kk];
      }
    }

    acc = __builtin_amdgcn_wmma_f32_16x16x32_bf16(false, av, false, bv,
                                                  (short)0, acc, false, false);
  }

  // epilogue: C/D layout VGPR r -> M = r + 8*(lane>=16), N = lane&15
  const float S = doS ? *Sp : 1.0f;
#pragma unroll
  for (int r = 0; r < 8; ++r) {
    int co = mBase + r + (kgrp << 3);
    float v = acc[r] + bias[co];
    if (doRelu) v = fmaxf(v, 0.f);
    v *= S;
    size_t oidx = ((size_t)bIdx * Cout + co) * LoutP + HALO + n;
    if (resid) v += (float)resid[oidx];
    out[oidx] = BF16(v);
  }
}

// ---- MaxPool1d(k=3,s=2,p=1) * S on halo-padded rows -----------------------
// Input is post-ReLU (>=0), so reading the zero halo never changes the max.
__global__ void maxpool3s2_kernel(const __bf16* __restrict__ in, __bf16* __restrict__ out,
                                  long BC, int LinP, int LoutP, int Lout,
                                  const float* __restrict__ Sp) {
  long i = (long)blockIdx.x * blockDim.x + threadIdx.x;
  if (i >= BC * (long)Lout) return;
  int  l  = (int)(i % Lout);
  long bc = i / Lout;
  const __bf16* p = in + bc * (size_t)LinP + HALO;
  float m = fmaxf(fmaxf((float)p[2 * l - 1], (float)p[2 * l]), (float)p[2 * l + 1]);
  out[bc * (size_t)LoutP + HALO + l] = BF16(m * (*Sp));
}

// ---- head pooling: feat[b,:] = [ S*max_L , S*mean_L ] ---------------------
__global__ void head_pool_kernel(const __bf16* __restrict__ in, float* __restrict__ feat,
                                 int C, int L, int Lp, const float* __restrict__ Sp) {
  int i = blockIdx.x * blockDim.x + threadIdx.x;
  if (i >= 64 * C) return;
  int b = i / C, c = i % C;
  const __bf16* p = in + ((size_t)b * C + c) * Lp + HALO;
  float mx = -3.4e38f, sum = 0.f;
  for (int l = 0; l < L; ++l) {
    float v = (float)p[l];
    mx = fmaxf(mx, v);
    sum += v;
  }
  float S = *Sp;
  feat[(size_t)b * 2 * C + c]     = S * mx;
  feat[(size_t)b * 2 * C + C + c] = S * (sum / (float)L);
}

// ---- head linear + logits BN affine ---------------------------------------
__global__ void head_linear_kernel(const float* __restrict__ feat, const float* __restrict__ lw,
                                   const float* __restrict__ lb, const float* __restrict__ hs,
                                   const float* __restrict__ hh, float* __restrict__ outp,
                                   int D, int O) {
  int b = blockIdx.x;
  int o = threadIdx.x;
  if (o < O) {
    const float* f = feat + (size_t)b * D;
    const float* w = lw + (size_t)o * D;
    float acc = 0.f;
    for (int d = 0; d < D; ++d) acc += f[d] * w[d];
    outp[b * O + o] = (acc + lb[o]) * hs[o] + hh[o];
  }
}

// ---------------------------------------------------------------------------
extern "C" void kernel_launch(void* const* d_in, const int* in_sizes, int n_in,
                              void* d_out, int out_size, void* d_ws, size_t ws_size,
                              hipStream_t stream) {
  (void)in_sizes; (void)n_in; (void)out_size; (void)ws_size;

  // ---- walk inputs in setup_inputs() insertion order ----
  int idx = 0;
  auto nextf = [&]() { return (const float*)d_in[idx++]; };
  const float* x  = nextf();
  const float* ts = nextf();

  struct ConvIn { const float *w, *b, *g, *be, *mu, *va; int cout, cin, kw, kpad; };
  ConvIn cv[24]; int nc = 0;
  auto takeConv = [&](int cout, int cin, int kw) {
    ConvIn c;
    c.w = nextf(); c.b = nextf();
    c.g = nextf(); c.be = nextf(); c.mu = nextf(); c.va = nextf();  // gamma,beta,mean,var
    c.cout = cout; c.cin = cin; c.kw = kw; c.kpad = (cin * kw + 31) & ~31;
    cv[nc] = c; return nc++;
  };

  int stem = takeConv(64, 12, 7);
  static const int cfg[8][3] = {{64,64,1},{64,64,1},{64,128,2},{128,128,1},
                                {128,256,2},{256,256,1},{256,512,2},{512,512,1}};
  int c1i[8], c2i[8], dni[8];
  for (int i = 0; i < 8; ++i) {
    int cin = cfg[i][0], cout = cfg[i][1], st = cfg[i][2];
    c1i[i] = takeConv(cout, cin, 3);
    c2i[i] = takeConv(cout, cout, 3);
    dni[i] = (st != 1 || cin != cout) ? takeConv(cout, cin, 1) : -1;
  }
  const float* lw  = nextf(); const float* lb  = nextf();
  const float* hg  = nextf(); const float* hbe = nextf();
  const float* hmu = nextf(); const float* hva = nextf();

  // ---- workspace bump allocator ----
  char* wsb = (char*)d_ws; size_t off = 0;
  auto alloc = [&](size_t bytes) {
    off = (off + 255) & ~(size_t)255;
    void* p = wsb + off; off += bytes; return p;
  };

  const int B = 64;
  __bf16* xb = (__bf16*)alloc((size_t)B * 12 * (4096 + 16) * 2);
  __bf16* A0 = (__bf16*)alloc((size_t)B * 64 * (2048 + 16) * 2);   // stem / conv1 out
  __bf16* A1 = (__bf16*)alloc((size_t)B * 512 * (128 + 16) * 2 > (size_t)B * 64 * (1024 + 16) * 2
                              ? (size_t)B * 512 * (128 + 16) * 2
                              : (size_t)B * 64 * (1024 + 16) * 2); // block out (max shape)
  __bf16* A2 = (__bf16*)alloc((size_t)B * 512 * (128 + 16) * 2);   // downsample identity
  float*  Sp = (float*)alloc(sizeof(float));
  __bf16* wq[24]; float* bq[24];
  for (int i = 0; i < nc; ++i) {
    wq[i] = (__bf16*)alloc((size_t)cv[i].cout * cv[i].kpad * 2);
    bq[i] = (float*)alloc((size_t)cv[i].cout * 4);
  }
  float* lwf  = (float*)alloc(27 * 1024 * 4);
  float* lbf  = (float*)alloc(27 * 4);
  float* hsv  = (float*)alloc(27 * 4);
  float* hhv  = (float*)alloc(27 * 4);
  float* feat = (float*)alloc((size_t)B * 1024 * 4);

  auto zh = [&](__bf16* buf, long rows, int Lp) {
    long n = rows * 16;
    zero_halo_kernel<<<(int)((n + 255) / 256), 256, 0, stream>>>(buf, rows, Lp);
  };

  // templated conv dispatch over the statically known layer shapes
  auto launch_conv = [&](int cin, int kw, int linp, dim3 grid,
                         const __bf16* in_, __bf16* out_, const __bf16* w_,
                         const float* b_, const __bf16* r_,
                         int cout, int loutp, int st, int pd, int rl, int dS) {
#define CONV_CASE(CI, KK, LP)                                                    \
    if (cin == CI && kw == KK && linp == LP) {                                   \
      conv1d_wmma_kernel<CI, KK, LP><<<grid, 128, 0, stream>>>(                  \
          in_, out_, w_, b_, r_, cout, loutp, st, pd, rl, dS, Sp);               \
      return;                                                                    \
    }
    CONV_CASE(64, 3, 1040)  CONV_CASE(128, 3, 528)
    CONV_CASE(256, 3, 272)  CONV_CASE(512, 3, 144)
    CONV_CASE(64, 1, 1040)  CONV_CASE(128, 1, 528)  CONV_CASE(256, 1, 272)
#undef CONV_CASE
  };

  // ---- parameter fusion (tiny, per launch, deterministic) ----
  sum_ts_kernel<<<1, 32, 0, stream>>>(ts, Sp);
  for (int i = 0; i < nc; ++i) {
    int totp = cv[i].cout * cv[i].kpad;
    fuse_conv_kernel<<<(totp + 255) / 256, 256, 0, stream>>>(
        cv[i].w, cv[i].b, cv[i].g, cv[i].be, cv[i].mu, cv[i].va,
        ts, wq[i], bq[i], cv[i].cout, cv[i].cin, cv[i].kw, cv[i].kpad);
  }
  fuse_head_kernel<<<(27 * 1024 + 255) / 256, 256, 0, stream>>>(
      lw, lb, hg, hbe, hmu, hva, ts, lwf, lbf, hsv, hhv, 27, 1024);

  // ---- input to padded bf16 ----
  {
    long n = (long)B * 12 * 4096;
    cvt_pad_kernel<<<(int)((n + 255) / 256), 256, 0, stream>>>(x, xb, B * 12L, 4096, 4112);
    zh(xb, B * 12L, 4112);
  }

  // ---- stem: conv(12->64,k7,s2,p3)+BN+ReLU, then S*maxpool ----
  conv1d_wmma_kernel<12, 7, 4112><<<dim3(2048 / 64, 64 / 16, B), 128, 0, stream>>>(
      xb, A0, wq[stem], bq[stem], nullptr,
      64, 2064, /*stride*/2, /*pad*/3, /*relu*/1, /*S*/0, Sp);
  zh(A0, (long)B * 64, 2064);
  {
    long n = (long)B * 64 * 1024;
    maxpool3s2_kernel<<<(int)((n + 255) / 256), 256, 0, stream>>>(
        A0, A1, (long)B * 64, 2064, 1040, 1024, Sp);
    zh(A1, (long)B * 64, 1040);
  }

  // ---- residual blocks ----
  __bf16* cur = A1; int curL = 1024, curLp = 1040;
  for (int i = 0; i < 8; ++i) {
    int cin = cfg[i][0], cout = cfg[i][1], st = cfg[i][2];
    int Lo = curL / st, LoP = Lo + 16;
    const __bf16* ident = cur;
    if (dni[i] >= 0) {  // 1x1 downsample + BN (no relu, no S); halo never read
      launch_conv(cin, 1, curLp, dim3(Lo / 64, cout / 16, B),
                  cur, A2, wq[dni[i]], bq[dni[i]], nullptr, cout, LoP, st, 0, 0, 0);
      ident = A2;
    }
    // conv1 + BN + ReLU + *S
    launch_conv(cin, 3, curLp, dim3(Lo / 64, cout / 16, B),
                cur, A0, wq[c1i[i]], bq[c1i[i]], nullptr, cout, LoP, st, 1, 1, 1);
    zh(A0, (long)B * cout, LoP);
    // conv2 + BN + ReLU + *S + residual (element-wise in-place resid read)
    launch_conv(cout, 3, LoP, dim3(Lo / 64, cout / 16, B),
                A0, cur, wq[c2i[i]], bq[c2i[i]], ident, cout, LoP, 1, 1, 1, 1);
    zh(cur, (long)B * cout, LoP);
    curL = Lo; curLp = LoP;
  }

  // ---- head: [S*max, S*avg] pooling, linear, logits BN ----
  head_pool_kernel<<<(B * 512 + 255) / 256, 256, 0, stream>>>(cur, feat, 512, curL, curLp, Sp);
  head_linear_kernel<<<B, 32, 0, stream>>>(feat, lwf, lbf, hsv, hhv, (float*)d_out, 1024, 27);
}
